// GAT_res_15547781612260
// MI455X (gfx1250) — matrix operand order repaired and checked by reading, not compiled, not run on hardware
//
#include <hip/hip_runtime.h>

#define HEADS 6

typedef __attribute__((ext_vector_type(16))) __bf16    v16bf;
typedef __attribute__((ext_vector_type(8)))  float     v8f;
typedef __attribute__((ext_vector_type(4)))  unsigned  u32x4;
typedef __attribute__((ext_vector_type(8)))  unsigned  u32x8;

struct APair { u32x4 lo, hi; };   // 32B, trivially copyable -> bit_cast to v16bf

// ---------------- helpers ----------------

__device__ __forceinline__ unsigned short f32_to_bf16(float f) {
  unsigned u = __float_as_uint(f);
  u += 0x7FFFu + ((u >> 16) & 1u);          // round-to-nearest-even
  return (unsigned short)(u >> 16);
}

// order-preserving float<->uint map so atomicMax(u32) == float max
__device__ __forceinline__ unsigned ord_enc(float f) {
  unsigned u = __float_as_uint(f);
  return (u & 0x80000000u) ? ~u : (u | 0x80000000u);
}
__device__ __forceinline__ float ord_dec(unsigned u) {
  unsigned v = (u & 0x80000000u) ? (u & 0x7FFFFFFFu) : ~u;
  return __uint_as_float(v);
}

// ---------------- conversion / init kernels ----------------

// x[N,K] f32 (row stride K) -> xbf[N,Kp] bf16, zero padded
__global__ void k_cvt_x(const float* __restrict__ x, unsigned short* __restrict__ xb,
                        int K, int Kp, long n) {
  long idx = (long)blockIdx.x * blockDim.x + threadIdx.x;
  if (idx >= n) return;
  long row = idx / Kp;
  int  k   = (int)(idx % Kp);
  xb[idx] = (k < K) ? f32_to_bf16(x[row * K + k]) : (unsigned short)0;
}

// W[K,M] f32 -> Wt[Mp,Kp] bf16 (transposed, zero padded)
__global__ void k_cvt_wt(const float* __restrict__ w, unsigned short* __restrict__ wt,
                         int K, int M, int Kp, long n) {
  long idx = (long)blockIdx.x * blockDim.x + threadIdx.x;
  if (idx >= n) return;
  long m = idx / Kp;
  int  k = (int)(idx % Kp);
  wt[idx] = (m < M && k < K) ? f32_to_bf16(w[(long)k * M + m]) : (unsigned short)0;
}

__global__ void k_fill_u32(unsigned* __restrict__ p, unsigned v, long n) {
  long idx = (long)blockIdx.x * blockDim.x + threadIdx.x;
  if (idx < n) p[idx] = v;
}

// ---------------- WMMA GEMM:  C[N,Mp] f32 = A[N,Kp] bf16 * Bt[Mp,Kp]^T bf16 ----------------
// one wave -> 16 x 64 of C (4 adjacent 16x16 tiles sharing the A fragment),
// software-pipelined: next k-slab's A + 4xB fragments are loaded before the
// current slab's 4 WMMAs so loads overlap matrix math across iterations.
__global__ void k_gemm_wmma(const unsigned short* __restrict__ A,
                            const unsigned short* __restrict__ Bt,
                            float* __restrict__ C,
                            int Kp, int Mp, int Mtiles) {
  const int wave = threadIdx.x >> 5;
  const int lane = threadIdx.x & 31;
  const int ct0  = (blockIdx.x * 8 + wave) * 4;    // first of 4 column tiles (wave-uniform)
  if (ct0 >= Mtiles) return;
  const int rt   = blockIdx.y;                     // row tile (N assumed mult of 16)
  const int m    = rt * 16 + (lane & 15);
  const int hiA  = (lane >> 4) * 16;  // A: lanes 16-31 take K 16..31 half
  const int hiB  = (lane >> 4) * 16;  // B: lanes 16-31 take K 16..31 half

  const unsigned short* arow = A + (long)m * Kp + hiA;
  // clamp OOB tiles onto the last valid tile (garbage math, store predicated off)
  const int c1 = (ct0 + 1 < Mtiles) ? ct0 + 1 : Mtiles - 1;
  const int c2 = (ct0 + 2 < Mtiles) ? ct0 + 2 : Mtiles - 1;
  const int c3 = (ct0 + 3 < Mtiles) ? ct0 + 3 : Mtiles - 1;
  const unsigned short* b0 = Bt + (long)(ct0 * 16 + (lane & 15)) * Kp + hiB;
  const unsigned short* b1 = Bt + (long)(c1  * 16 + (lane & 15)) * Kp + hiB;
  const unsigned short* b2 = Bt + (long)(c2  * 16 + (lane & 15)) * Kp + hiB;
  const unsigned short* b3 = Bt + (long)(c3  * 16 + (lane & 15)) * Kp + hiB;

  // ---- prologue: load k-slab 0 ----
  APair ar;
  ar.lo = *(const u32x4*)(arow);
  ar.hi = *(const u32x4*)(arow + 8);
  u32x8 rb0 = *(const u32x8*)(b0);
  u32x8 rb1 = *(const u32x8*)(b1);
  u32x8 rb2 = *(const u32x8*)(b2);
  u32x8 rb3 = *(const u32x8*)(b3);

  v8f acc0 = {}, acc1 = {}, acc2 = {}, acc3 = {};
  for (int kb = 0; kb < Kp; kb += 32) {
    const int kn = (kb + 32 < Kp) ? kb + 32 : 0;   // clamp keeps loads in-bounds
    __builtin_prefetch(arow + kn + 64, 0, 1);      // global_prefetch_b8
    __builtin_prefetch(b0 + kn + 64, 0, 1);
    // issue next slab's loads before this slab's WMMAs
    APair an;
    an.lo = *(const u32x4*)(arow + kn);
    an.hi = *(const u32x4*)(arow + kn + 8);
    u32x8 n0 = *(const u32x8*)(b0 + kn);
    u32x8 n1 = *(const u32x8*)(b1 + kn);
    u32x8 n2 = *(const u32x8*)(b2 + kn);
    u32x8 n3 = *(const u32x8*)(b3 + kn);
    v16bf av = __builtin_bit_cast(v16bf, ar);
    acc0 = __builtin_amdgcn_wmma_f32_16x16x32_bf16(false, av, false,
              __builtin_bit_cast(v16bf, rb0), (short)0, acc0, false, false);
    acc1 = __builtin_amdgcn_wmma_f32_16x16x32_bf16(false, av, false,
              __builtin_bit_cast(v16bf, rb1), (short)0, acc1, false, false);
    acc2 = __builtin_amdgcn_wmma_f32_16x16x32_bf16(false, av, false,
              __builtin_bit_cast(v16bf, rb2), (short)0, acc2, false, false);
    acc3 = __builtin_amdgcn_wmma_f32_16x16x32_bf16(false, av, false,
              __builtin_bit_cast(v16bf, rb3), (short)0, acc3, false, false);
    ar = an; rb0 = n0; rb1 = n1; rb2 = n2; rb3 = n3;
  }
  // C/D layout: VGPR r holds row rt*16 + (lane>=16?8:0) + r, col = tile*16 + lane%16
  float* crow = C + (long)(rt * 16 + (lane >> 4) * 8) * Mp + (lane & 15);
#pragma unroll
  for (int r = 0; r < 8; ++r) {
    float* p = crow + (long)r * Mp;
    p[ct0 * 16] = acc0[r];
    if (ct0 + 1 < Mtiles) p[(ct0 + 1) * 16] = acc1[r];
    if (ct0 + 2 < Mtiles) p[(ct0 + 2) * 16] = acc2[r];
    if (ct0 + 3 < Mtiles) p[(ct0 + 3) * 16] = acc3[r];
  }
}

// ---------------- attention ----------------

// a_src[n,h] = dot(xp[n,h,:], att_src[h,:]); same for a_dst. one wave per (n,h)
__global__ void k_attn_dots(const float* __restrict__ xp,
                            const float* __restrict__ as_, const float* __restrict__ ad_,
                            float* __restrict__ o_s, float* __restrict__ o_d,
                            int N, int cout, int Mp) {
  long tid = (long)blockIdx.x * blockDim.x + threadIdx.x;
  long wid = tid >> 5;
  int lane = threadIdx.x & 31;
  if (wid >= (long)N * HEADS) return;
  int n = (int)(wid / HEADS), h = (int)(wid % HEADS);
  const float* row = xp  + (long)n * Mp + (long)h * cout;
  const float* vs  = as_ + (long)h * cout;
  const float* vd  = ad_ + (long)h * cout;
  float s = 0.f, d = 0.f;
  for (int c = lane; c < cout; c += 32) { float v = row[c]; s += v * vs[c]; d += v * vd[c]; }
  for (int o = 16; o > 0; o >>= 1) { s += __shfl_xor(s, o, 32); d += __shfl_xor(d, o, 32); }
  if (lane == 0) { o_s[wid] = s; o_d[wid] = d; }
}

// e = leaky_relu(a_src[s]+a_dst[d]); store logit; segment-max into emax (ordered u32)
__global__ void k_edge_max(const float* __restrict__ a_src, const float* __restrict__ a_dst,
                           const int* __restrict__ srcA, const int* __restrict__ dstA,
                           int E, int N, float* __restrict__ eval, unsigned* __restrict__ emax) {
  long idx = (long)blockIdx.x * blockDim.x + threadIdx.x;
  long total = ((long)E + N) * HEADS;
  if (idx >= total) return;
  long e = idx / HEADS; int h = (int)(idx % HEADS);
  int s, d;
  if (e < E) { s = srcA[e]; d = dstA[e]; } else { s = d = (int)(e - E); }  // self loop
  float v = a_src[(long)s * HEADS + h] + a_dst[(long)d * HEADS + h];
  v = v > 0.f ? v : 0.2f * v;
  eval[idx] = v;
  atomicMax(&emax[(long)d * HEADS + h], ord_enc(v));
}

// alpha = exp(e - emax[d]); segment-sum into denom
__global__ void k_edge_exp(const int* __restrict__ dstA, int E, int N,
                           float* __restrict__ alpha, const unsigned* __restrict__ emax,
                           float* __restrict__ denom) {
  long idx = (long)blockIdx.x * blockDim.x + threadIdx.x;
  long total = ((long)E + N) * HEADS;
  if (idx >= total) return;
  long e = idx / HEADS; int h = (int)(idx % HEADS);
  int d = (e < E) ? dstA[e] : (int)(e - E);
  float v = __expf(alpha[idx] - ord_dec(emax[(long)d * HEADS + h]));
  alpha[idx] = v;
  atomicAdd(&denom[(long)d * HEADS + h], v);
}

// accum[d,:] += (alpha/denom[d]) * xp[s,h,:]; one wave per (edge, head)
__global__ void k_edge_scatter(const float* __restrict__ xp, const float* __restrict__ alpha,
                               const float* __restrict__ denom,
                               const int* __restrict__ srcA, const int* __restrict__ dstA,
                               int E, int N, int cout, int Mp, float* __restrict__ accum) {
  long tid = (long)blockIdx.x * blockDim.x + threadIdx.x;
  long wid = tid >> 5;
  int lane = threadIdx.x & 31;
  long total = ((long)E + N) * HEADS;
  if (wid >= total) return;
  long e = wid / HEADS; int h = (int)(wid % HEADS);
  int s, d;
  if (e < E) { s = srcA[e]; d = dstA[e]; } else { s = d = (int)(e - E); }
  float a = alpha[wid] / denom[(long)d * HEADS + h];
  const float* src_row = xp + (long)s * Mp + (long)h * cout;
  float* dst_row = accum + (long)d * cout;
  for (int c = lane; c < cout; c += 32) atomicAdd(&dst_row[c], a * src_row[c]);
}

// out = accum/HEADS + bias
__global__ void k_finalize(const float* __restrict__ accum, const float* __restrict__ bias,
                           float* __restrict__ out, int cout, long n) {
  long idx = (long)blockIdx.x * blockDim.x + threadIdx.x;
  if (idx >= n) return;
  int c = (int)(idx % cout);
  out[idx] = accum[idx] * (1.0f / HEADS) + bias[c];
}

__global__ void k_add2(const float* __restrict__ a, const float* __restrict__ b,
                       float* __restrict__ o, long n) {
  long idx = (long)blockIdx.x * blockDim.x + threadIdx.x;
  if (idx < n) o[idx] = a[idx] + b[idx];
}

// final = x + relu(h), gather row ptr[b+1]-1 for each graph
__global__ void k_final_gather(const float* __restrict__ x, const float* __restrict__ h,
                               const int* __restrict__ gptr, float* __restrict__ out, int C) {
  int b = blockIdx.x;
  int row = gptr[b + 1] - 1;
  for (int c = threadIdx.x; c < C; c += blockDim.x) {
    float v = h[(long)row * C + c];
    out[(long)b * C + c] = x[(long)row * C + c] + (v > 0.f ? v : 0.f);
  }
}

// ---------------- host orchestration ----------------

struct Scratch {
  unsigned short* xbf; unsigned short* wtb;
  float* xp; float* a_src; float* a_dst;
  unsigned* emax; float* denom; float* alpha; float* accum;
};

static inline long cdiv(long a, long b) { return (a + b - 1) / b; }

static void run_conv(const float* xin, const float* W, const float* att_s,
                     const float* att_d, const float* bias, int K, int cout, float* out,
                     const Scratch& S, const int* srcA, const int* dstA,
                     int E, int N, hipStream_t stream) {
  const int  M  = HEADS * cout;
  const int  Kp = ((K + 31) / 32) * 32;
  const int  Mp = ((M + 15) / 16) * 16;
  const int  Mtiles = Mp / 16;
  const long ET = (long)E + N;

  { long n = (long)N * Kp;
    k_cvt_x<<<cdiv(n, 256), 256, 0, stream>>>(xin, S.xbf, K, Kp, n); }
  { long n = (long)Mp * Kp;
    k_cvt_wt<<<cdiv(n, 256), 256, 0, stream>>>(W, S.wtb, K, M, Kp, n); }
  { dim3 g((unsigned)cdiv(Mtiles, 32), (unsigned)(N / 16));  // 8 waves x 4 tiles per block
    k_gemm_wmma<<<g, 256, 0, stream>>>(S.xbf, S.wtb, S.xp, Kp, Mp, Mtiles); }
  { long n = (long)N * HEADS * 32;
    k_attn_dots<<<cdiv(n, 256), 256, 0, stream>>>(S.xp, att_s, att_d, S.a_src, S.a_dst,
                                                  N, cout, Mp); }
  { long n = (long)N * HEADS;
    k_fill_u32<<<cdiv(n, 256), 256, 0, stream>>>(S.emax, 0x007FFFFFu, n);   // ord(-inf)
    k_fill_u32<<<cdiv(n, 256), 256, 0, stream>>>((unsigned*)S.denom, 0u, n); }
  { long n = (long)N * cout;
    k_fill_u32<<<cdiv(n, 256), 256, 0, stream>>>((unsigned*)S.accum, 0u, n); }
  { long n = ET * HEADS;
    k_edge_max<<<cdiv(n, 256), 256, 0, stream>>>(S.a_src, S.a_dst, srcA, dstA, E, N,
                                                 S.alpha, S.emax);
    k_edge_exp<<<cdiv(n, 256), 256, 0, stream>>>(dstA, E, N, S.alpha, S.emax, S.denom); }
  { long n = ET * HEADS * 32;
    k_edge_scatter<<<cdiv(n, 256), 256, 0, stream>>>(S.xp, S.alpha, S.denom, srcA, dstA,
                                                     E, N, cout, Mp, S.accum); }
  { long n = (long)N * cout;
    k_finalize<<<cdiv(n, 256), 256, 0, stream>>>(S.accum, bias, out, cout, n); }
}

extern "C" void kernel_launch(void* const* d_in, const int* in_sizes, int n_in,
                              void* d_out, int out_size, void* d_ws, size_t ws_size,
                              hipStream_t stream) {
  const int DIMS[4] = {1028, 128, 256, 1028};
  const int N = in_sizes[0] / DIMS[0];         // 20000
  const int E = in_sizes[1] / 2;               // 240000

  const float* x    = (const float*)d_in[0];
  const int*   eidx = (const int*)d_in[1];
  const int*   gptr = (const int*)d_in[2];
  const int*   srcA = eidx;
  const int*   dstA = eidx + E;

  // ---- workspace layout (256B aligned slices of d_ws) ----
  const int KPMAX = 1056, MPMAX = 6176, CMAX = 1028;
  char*  base = (char*)d_ws;
  size_t off  = 0;
  auto take = [&](size_t bytes) -> char* {
    char* p = base + off;
    off += (bytes + 255) & ~(size_t)255;
    return p;
  };
  Scratch S;
  S.xbf   = (unsigned short*)take((size_t)N * KPMAX * 2);
  S.wtb   = (unsigned short*)take((size_t)MPMAX * KPMAX * 2);
  S.xp    = (float*)take((size_t)N * MPMAX * 4);
  S.a_src = (float*)take((size_t)N * HEADS * 4);
  S.a_dst = (float*)take((size_t)N * HEADS * 4);
  S.emax  = (unsigned*)take((size_t)N * HEADS * 4);
  S.denom = (float*)take((size_t)N * HEADS * 4);
  S.alpha = (float*)take((size_t)(E + N) * HEADS * 4);
  S.accum = (float*)take((size_t)N * CMAX * 4);
  float* t1 = (float*)take((size_t)N * CMAX * 4);
  float* t2 = (float*)take((size_t)N * CMAX * 4);
  float* tr = (float*)take((size_t)N * CMAX * 4);
  float* hb = (float*)take((size_t)N * CMAX * 4);
  (void)ws_size; (void)n_in; (void)out_size;

  const float* bin = x;
  for (int b = 0; b < 3; ++b) {
    const int cin  = DIMS[b];
    const int cblk = DIMS[b + 1];
    const float* P[3][4];
    for (int c = 0; c < 3; ++c)
      for (int j = 0; j < 4; ++j)
        P[c][j] = (const float*)d_in[3 + (b * 3 + c) * 4 + j];
    // c1: cin -> cin ; c2: cin -> cblk ; cr: cin -> cblk
    run_conv(bin, P[0][0], P[0][1], P[0][2], P[0][3], cin, cin,  t1, S, srcA, dstA, E, N, stream);
    run_conv(t1,  P[1][0], P[1][1], P[1][2], P[1][3], cin, cblk, t2, S, srcA, dstA, E, N, stream);
    run_conv(bin, P[2][0], P[2][1], P[2][2], P[2][3], cin, cblk, tr, S, srcA, dstA, E, N, stream);
    { long n = (long)N * cblk;
      k_add2<<<cdiv(n, 256), 256, 0, stream>>>(t2, tr, hb, n); }
    bin = hb;
  }
  k_final_gather<<<8, 256, 0, stream>>>(x, hb, gptr, (float*)d_out, DIMS[0]);
}
